// TransNet_23261542875481
// MI455X (gfx1250) — matrix operand order repaired and checked by reading, not compiled, not run on hardware
//
#include <hip/hip_runtime.h>
#include <hip/hip_bf16.h>
#include <math.h>

typedef _Float16 half_t;
typedef __attribute__((ext_vector_type(16))) _Float16 v16h;
typedef __attribute__((ext_vector_type(8)))  _Float16 v8h;
typedef __attribute__((ext_vector_type(8)))  float    v8f;
typedef __attribute__((ext_vector_type(4)))  int      v4i;

#define EPSV 1e-5f

#if defined(__gfx1250__) && __has_builtin(__builtin_amdgcn_global_load_async_to_lds_b128)
#define HAVE_ASYNC_LDS 1
typedef __attribute__((address_space(1))) v4i* gv4ip_t;   // global int4*
typedef __attribute__((address_space(3))) v4i* lv4ip_t;   // LDS int4*
#else
#define HAVE_ASYNC_LDS 0
#endif

// ---------- wave32 reduction ----------
__device__ __forceinline__ float wave_sum(float v) {
#pragma unroll
  for (int m = 16; m > 0; m >>= 1) v += __shfl_xor(v, m, 32);
  return v;
}

// ---------- WMMA fragment loaders (gfx1250 16x16x32 f16 layouts) ----------
// A (MxK = 16x32): lane = M row; lanes 0-15 hold K {0..7,16..23}, lanes 16-31 K {8..15,24..31}
__device__ __forceinline__ v16h load_a_frag(const half_t* __restrict__ A, int lda,
                                            int row, int k0, int lane) {
  const half_t* p = A + (size_t)row * lda + k0 + ((lane & 16) ? 8 : 0);
  v8h lo = *(const v8h*)p;
  v8h hi = *(const v8h*)(p + 16);
  v16h f;
#pragma unroll
  for (int i = 0; i < 8; i++) { f[i] = lo[i]; f[i + 8] = hi[i]; }
  return f;
}
// B (KxN = 32x16) built from row-major W (N x K): lane = N row of W; lanes 0-15 K {0..7,16..23}, 16-31 K {8..15,24..31}
__device__ __forceinline__ v16h load_b_frag(const half_t* __restrict__ W, int ldw,
                                            int row, int k0, int lane) {
  const half_t* p = W + (size_t)row * ldw + k0 + ((lane & 16) ? 8 : 0);
  v8h lo = *(const v8h*)p;
  v8h hi = *(const v8h*)(p + 16);
  v16h f;
#pragma unroll
  for (int i = 0; i < 8; i++) { f[i] = lo[i]; f[i + 8] = hi[i]; }
  return f;
}

// ---------- weight/activation prep ----------
// W_eff[o][tau][c] = s2[o] * sum_i sw[o,i,c] * s1[i] * tw_group(i)[tau]   (f16)
__global__ void k_prep_weff(const float* __restrict__ sw, const float* __restrict__ bn1_g,
                            const float* __restrict__ bn2_g,
                            const float* __restrict__ tw1, const float* __restrict__ tw2,
                            const float* __restrict__ tw3, const float* __restrict__ tw4,
                            half_t* __restrict__ Weff) {
  int id = blockIdx.x * blockDim.x + threadIdx.x;   // 128*65*64
  if (id >= 128 * 65 * 64) return;
  int o = id / (65 * 64);
  int tau = (id / 64) % 65;
  int c = id % 64;
  const float rs = rsqrtf(1.f + EPSV);
  const int Kg[4] = {15, 25, 51, 65};
  const int Pg[4] = {7, 12, 25, 32};
  const float* tws[4] = {tw1, tw2, tw3, tw4};
  float acc = 0.f;
  for (int g = 0; g < 4; g++) {
    int k = tau - 32 + Pg[g];
    if (k < 0 || k >= Kg[g]) continue;
    const float* tw = tws[g];
    for (int j = 0; j < 32; j++) {
      int i = g * 32 + j;
      acc += sw[((size_t)o * 128 + i) * 64 + c] * (bn1_g[i] * rs) * tw[j * Kg[g] + k];
    }
  }
  acc *= bn2_g[o] * rs;
  Weff[((size_t)o * 65 + tau) * 64 + c] = (half_t)acc;
}

// beta[o] = s2[o]*(sum_i sum_c sw[o,i,c]*(s1[i]*tb[i]+bn1_b[i]) + sb[o]) + bn2_b[o]
__global__ void k_prep_beta(const float* __restrict__ sw, const float* __restrict__ bn1_g,
                            const float* __restrict__ bn1_b,
                            const float* __restrict__ tb1, const float* __restrict__ tb2,
                            const float* __restrict__ tb3, const float* __restrict__ tb4,
                            const float* __restrict__ sb, const float* __restrict__ bn2_g,
                            const float* __restrict__ bn2_b, float* __restrict__ beta) {
  int o = threadIdx.x;   // 128 threads
  const float rs = rsqrtf(1.f + EPSV);
  float acc = sb[o];
  for (int i = 0; i < 128; i++) {
    int g = i >> 5, j = i & 31;
    float tb = (g == 0) ? tb1[j] : (g == 1) ? tb2[j] : (g == 2) ? tb3[j] : tb4[j];
    float bi = bn1_g[i] * rs * tb + bn1_b[i];
    float swsum = 0.f;
    for (int c = 0; c < 64; c++) swsum += sw[((size_t)o * 128 + i) * 64 + c];
    acc += swsum * bi;
  }
  beta[o] = bn2_g[o] * rs * acc + bn2_b[o];
}

// x (16,64,1000) f32 -> xT (16, 1072, 64) f16, time padded +-32 with zeros, t-major rows
__global__ void k_prep_x(const float* __restrict__ x, half_t* __restrict__ xT) {
  int id = blockIdx.x * blockDim.x + threadIdx.x;   // 16*1072*64
  if (id >= 16 * 1072 * 64) return;
  int b = id / (1072 * 64);
  int t = (id / 64) % 1072;
  int c = id % 64;
  int ts = t - 32;
  float v = (ts >= 0 && ts < 1000) ? x[((size_t)b * 64 + c) * 1000 + ts] : 0.f;
  xT[id] = (half_t)v;
}

__global__ void k_cast_f16(const float* __restrict__ src, half_t* __restrict__ dst, int n) {
  int id = blockIdx.x * blockDim.x + threadIdx.x;
  if (id < n) dst[id] = (half_t)src[id];
}

// ---------- fused temporal+spatial conv + BN2 + ELU, via WMMA ----------
// h[b,o,t] = ELU( sum_tau sum_c Weff[o,tau,c] * xT[b, t+tau, c] + beta[o] )
// One wave = 16 (o) x 16 (t) output tile. 65 taus x 2 K-chunks = 130 WMMAs/wave.
__global__ void k_conv_wmma(const half_t* __restrict__ Weff, const half_t* __restrict__ xT,
                            const float* __restrict__ beta, float* __restrict__ h) {
  int lane = threadIdx.x & 31;
  int w = blockIdx.x * 8 + (threadIdx.x >> 5);      // 8064 waves: 16 b * 8 o-tiles * 63 t-tiles
  int b = w / (8 * 63);
  int rem = w % (8 * 63);
  int o0 = (rem / 63) << 4;
  int t0 = (rem % 63) << 4;
  v8f acc = {};
  int arow = o0 + (lane & 15);
  const half_t* xb = xT + (size_t)b * 1072 * 64;
  for (int tau = 0; tau < 65; ++tau) {
    int brow = t0 + tau + (lane & 15);
    int kbase = tau * 64;
    __builtin_prefetch(xb + (size_t)(brow + 8) * 64, 0, 1);   // global_prefetch_b8
    v16h a0 = load_a_frag(Weff, 65 * 64, arow, kbase, lane);
    v16h b0 = load_b_frag(xb, 64, brow, 0, lane);
    acc = __builtin_amdgcn_wmma_f32_16x16x32_f16(false, a0, false, b0, (short)0, acc, false, false);
    v16h a1 = load_a_frag(Weff, 65 * 64, arow, kbase + 32, lane);
    v16h b1 = load_b_frag(xb, 64, brow, 32, lane);
    acc = __builtin_amdgcn_wmma_f32_16x16x32_f16(false, a1, false, b1, (short)0, acc, false, false);
  }
  int t = t0 + (lane & 15);
  int obase = o0 + ((lane >> 4) << 3);
  if (t < 1000) {
#pragma unroll
    for (int v = 0; v < 8; v++) {
      int o = obase + v;
      float val = acc[v] + beta[o];
      val = val > 0.f ? val : expm1f(val);   // ELU(alpha=1)
      h[((size_t)b * 128 + o) * 1000 + t] = val;
    }
  }
}

// ---------- pooling: mean + log(clip(var,ddof=1)) over 64 windows of 50, stride 15 ----------
__global__ void k_pool(const float* __restrict__ h, float* __restrict__ X) {
  int id = blockIdx.x * blockDim.x + threadIdx.x;   // 16*128*64
  int b = id / (128 * 64);
  int ch = (id / 64) % 128;
  int wv = id % 64;
  const float* p = h + ((size_t)b * 128 + ch) * 1000 + wv * 15;
  float s = 0.f;
  for (int j = 0; j < 50; j++) s += p[j];
  float mean = s * (1.f / 50.f);
  float q = 0.f;
  for (int j = 0; j < 50; j++) { float d = p[j] - mean; q += d * d; }
  float var = q * (1.f / 49.f);
  float vl = logf(fminf(fmaxf(var, 1e-6f), 1e6f));
  X[((size_t)(b * 64 + wv)) * 128 + ch] = mean;
  X[((size_t)(1024 + b * 64 + wv)) * 128 + ch] = vl;
}

// ---------- layernorm (d=128), one wave per row, writes f16 ----------
__global__ void k_layernorm(const float* __restrict__ X, const float* __restrict__ g,
                            const float* __restrict__ bta, half_t* __restrict__ H) {
  int lane = threadIdx.x & 31;
  int r = blockIdx.x * 8 + (threadIdx.x >> 5);
  const float* xr = X + (size_t)r * 128;
  float v0[4];
  float s = 0.f;
#pragma unroll
  for (int j = 0; j < 4; j++) { v0[j] = xr[lane * 4 + j]; s += v0[j]; }
  float mean = wave_sum(s) * (1.f / 128.f);
  float q = 0.f;
#pragma unroll
  for (int j = 0; j < 4; j++) { float d = v0[j] - mean; q += d * d; }
  float var = wave_sum(q) * (1.f / 128.f);
  float inv = rsqrtf(var + EPSV);
  half_t* hr = H + (size_t)r * 128;
#pragma unroll
  for (int j = 0; j < 4; j++) {
    int c = lane * 4 + j;
    hr[c] = (half_t)((v0[j] - mean) * inv * g[c] + bta[c]);
  }
}

// ---------- generic WMMA GEMM: C(MxN) = A(MxK,f16) @ W(NxK,f16)^T + bias ----------
// Block = 8 waves sharing one m-row of tiles; the 16xK A panel (contiguous, lda==K) is
// staged once in LDS, via GLOBAL_LOAD_ASYNC_TO_LDS_B128 (ASYNCcnt) when available.
// epi: 0 = f32 store, 1 = f32 residual-add store, 2 = f16 store, 3 = f16 exact-GELU store
__global__ void k_gemm_wmma(const half_t* __restrict__ A, const half_t* __restrict__ W,
                            const float* __restrict__ bias, int M, int N, int K,
                            float* __restrict__ Cf, half_t* __restrict__ Ch,
                            const float* __restrict__ Res, int epi) {
  __shared__ half_t As[16 * 512];                 // max K = 512 -> 16KB
  int lane = threadIdx.x & 31;
  int wave = threadIdx.x >> 5;
  int ntn = N >> 4;
  int tile0 = blockIdx.x * 8;                     // 8 tiles/block, never crosses an m-row
  int m0 = (tile0 / ntn) << 4;                    // uniform across block
  int n0 = ((tile0 % ntn) + wave) << 4;

  // stage A panel rows m0..m0+15 (contiguous 16*K halves) into LDS
  const half_t* Ap = A + (size_t)m0 * K;
  int panel = 16 * K;                             // halves
  for (int ofs = threadIdx.x * 8; ofs < panel; ofs += 256 * 8) {
#if HAVE_ASYNC_LDS
    half_t* src = const_cast<half_t*>(Ap + ofs);
    __builtin_amdgcn_global_load_async_to_lds_b128((gv4ip_t)src, (lv4ip_t)(As + ofs), 0, 0);
#else
    *(v8h*)(As + ofs) = *(const v8h*)(Ap + ofs);
#endif
  }
#if HAVE_ASYNC_LDS
#if __has_builtin(__builtin_amdgcn_s_wait_asynccnt)
  __builtin_amdgcn_s_wait_asynccnt(0);
#else
  asm volatile("s_wait_asynccnt 0x0" ::: "memory");
#endif
#endif
  __syncthreads();

  v8f acc = {};
  int arow = lane & 15;                           // row within LDS panel
  int brow = n0 + (lane & 15);
  for (int k0 = 0; k0 < K; k0 += 32) {
    v16h af = load_a_frag(As, K, arow, k0, lane); // ds_load path
    v16h bf = load_b_frag(W, K, brow, k0, lane);  // L2-resident weights
    acc = __builtin_amdgcn_wmma_f32_16x16x32_f16(false, af, false, bf, (short)0, acc, false, false);
  }
  int col = n0 + (lane & 15);
  int mbase = m0 + ((lane >> 4) << 3);
  float bv = bias ? bias[col] : 0.f;
#pragma unroll
  for (int v = 0; v < 8; v++) {
    int row = mbase + v;
    float val = acc[v] + bv;
    if (epi == 1) val += Res[(size_t)row * N + col];
    if (epi == 3) val = 0.5f * val * (1.f + erff(val * 0.70710678118f));
    if (epi <= 1) Cf[(size_t)row * N + col] = val;
    else          Ch[(size_t)row * N + col] = (half_t)val;
  }
}

// ---------- attention: per (stream,b,head) block, n=64, dk=16 ----------
__global__ void k_attention(const float* __restrict__ Q, const float* __restrict__ K,
                            const float* __restrict__ V, half_t* __restrict__ O) {
  __shared__ float Ks[64][16];
  __shared__ float Vs[64][16];
  int tid = threadIdx.x;            // query row 0..63
  int pair = blockIdx.x >> 3;       // (stream,b) 0..31
  int head = blockIdx.x & 7;
  int r0 = pair * 64;
  int c0 = head * 16;
#pragma unroll
  for (int d = 0; d < 16; d++) {
    Ks[tid][d] = K[((size_t)(r0 + tid)) * 128 + c0 + d];
    Vs[tid][d] = V[((size_t)(r0 + tid)) * 128 + c0 + d];
  }
  __syncthreads();
  float q[16];
#pragma unroll
  for (int d = 0; d < 16; d++) q[d] = Q[((size_t)(r0 + tid)) * 128 + c0 + d];
  float sc[64];
  float mx = -1e30f;
  for (int kk = 0; kk < 64; kk++) {
    float s = 0.f;
#pragma unroll
    for (int d = 0; d < 16; d++) s += q[d] * Ks[kk][d];
    s *= 0.25f;                     // 1/sqrt(16)
    sc[kk] = s;
    mx = fmaxf(mx, s);
  }
  float denom = 0.f;
  for (int kk = 0; kk < 64; kk++) { sc[kk] = expf(sc[kk] - mx); denom += sc[kk]; }
  float invd = 1.f / denom;
  float out[16] = {};
  for (int kk = 0; kk < 64; kk++) {
    float p = sc[kk];
#pragma unroll
    for (int d = 0; d < 16; d++) out[d] += p * Vs[kk][d];
  }
#pragma unroll
  for (int d = 0; d < 16; d++)
    O[((size_t)(r0 + tid)) * 128 + c0 + d] = (half_t)(out[d] * invd);
}

// ---------- final merge conv (2x1) + BN3 + ELU ----------
__global__ void k_final_feat(const float* __restrict__ X, const float* __restrict__ ew,
                             const float* __restrict__ eb, const float* __restrict__ bn3_g,
                             const float* __restrict__ bn3_b, float* __restrict__ feat) {
  int id = blockIdx.x * blockDim.x + threadIdx.x;   // 16*64*128
  int b = id / (64 * 128);
  int on = (id / 128) % 64;
  int d = id % 128;
  float acc = eb[on];
  for (int i = 0; i < 64; i++) {
    float w0 = ew[((size_t)on * 64 + i) * 2 + 0];
    float w1 = ew[((size_t)on * 64 + i) * 2 + 1];
    acc += w0 * X[((size_t)(b * 64 + i)) * 128 + d]
         + w1 * X[((size_t)(1024 + b * 64 + i)) * 128 + d];
  }
  const float rs = rsqrtf(1.f + EPSV);
  float val = bn3_g[on] * rs * acc + bn3_b[on];
  val = val > 0.f ? val : expm1f(val);
  feat[((size_t)b * 64 + on) * 128 + d] = val;
}

// ---------- classifier ----------
__global__ void k_classifier(const float* __restrict__ feat, const float* __restrict__ cw,
                             const float* __restrict__ cb, float* __restrict__ out) {
  int lane = threadIdx.x;           // 32 threads
  int b = blockIdx.x >> 2;
  int cls = blockIdx.x & 3;
  const float* f = feat + (size_t)b * 8192;
  const float* w = cw + (size_t)cls * 8192;
  float s = 0.f;
  for (int j = lane; j < 8192; j += 32) s += f[j] * w[j];
  s = wave_sum(s);
  if (lane == 0) out[b * 4 + cls] = s + cb[cls];
}

extern "C" void kernel_launch(void* const* d_in, const int* in_sizes, int n_in,
                              void* d_out, int out_size, void* d_ws, size_t ws_size,
                              hipStream_t stream) {
  (void)in_sizes; (void)n_in; (void)out_size; (void)ws_size;
  const float* x     = (const float*)d_in[0];
  const float* tw1   = (const float*)d_in[1];
  const float* tb1   = (const float*)d_in[2];
  const float* tw2   = (const float*)d_in[3];
  const float* tb2   = (const float*)d_in[4];
  const float* tw3   = (const float*)d_in[5];
  const float* tb3   = (const float*)d_in[6];
  const float* tw4   = (const float*)d_in[7];
  const float* tb4   = (const float*)d_in[8];
  const float* bn1_g = (const float*)d_in[9];
  const float* bn1_b = (const float*)d_in[10];
  const float* sw    = (const float*)d_in[11];
  const float* sb    = (const float*)d_in[12];
  const float* bn2_g = (const float*)d_in[13];
  const float* bn2_b = (const float*)d_in[14];
  const float* ln1_g = (const float*)d_in[15];
  const float* ln1_b = (const float*)d_in[16];
  const float* ln2_g = (const float*)d_in[17];
  const float* ln2_b = (const float*)d_in[18];
  const float* wq    = (const float*)d_in[19];
  const float* bq    = (const float*)d_in[20];
  const float* wk    = (const float*)d_in[21];
  const float* bk    = (const float*)d_in[22];
  const float* wv    = (const float*)d_in[23];
  const float* bv    = (const float*)d_in[24];
  const float* wo    = (const float*)d_in[25];
  const float* bo    = (const float*)d_in[26];
  const float* f1w   = (const float*)d_in[27];
  const float* f1b   = (const float*)d_in[28];
  const float* f2w   = (const float*)d_in[29];
  const float* f2b   = (const float*)d_in[30];
  const float* ew    = (const float*)d_in[31];
  const float* eb    = (const float*)d_in[32];
  const float* bn3_g = (const float*)d_in[33];
  const float* bn3_b = (const float*)d_in[34];
  const float* cw    = (const float*)d_in[35];
  const float* cb    = (const float*)d_in[36];

  char* ws = (char*)d_ws;
  size_t off = 0;
  auto carve = [&](size_t bytes) -> void* {
    void* p = (void*)(ws + off);
    off = (off + bytes + 255) & ~(size_t)255;
    return p;
  };
  half_t* Weff = (half_t*)carve((size_t)128 * 65 * 64 * sizeof(half_t));
  float*  beta = (float*) carve(128 * sizeof(float));
  half_t* xT   = (half_t*)carve((size_t)16 * 1072 * 64 * sizeof(half_t));
  float*  h    = (float*) carve((size_t)16 * 128 * 1000 * sizeof(float));
  float*  X    = (float*) carve((size_t)2048 * 128 * sizeof(float));
  half_t* H    = (half_t*)carve((size_t)2048 * 128 * sizeof(half_t));
  float*  Qf   = (float*) carve((size_t)2048 * 128 * sizeof(float));
  float*  Kf   = (float*) carve((size_t)2048 * 128 * sizeof(float));
  float*  Vf   = (float*) carve((size_t)2048 * 128 * sizeof(float));
  half_t* Oh   = (half_t*)carve((size_t)2048 * 128 * sizeof(half_t));
  half_t* F1h  = (half_t*)carve((size_t)2048 * 512 * sizeof(half_t));
  half_t* wqh  = (half_t*)carve((size_t)4 * 128 * 128 * sizeof(half_t));
  half_t* wkh  = (half_t*)carve((size_t)4 * 128 * 128 * sizeof(half_t));
  half_t* wvh  = (half_t*)carve((size_t)4 * 128 * 128 * sizeof(half_t));
  half_t* woh  = (half_t*)carve((size_t)4 * 128 * 128 * sizeof(half_t));
  half_t* f1wh = (half_t*)carve((size_t)4 * 512 * 128 * sizeof(half_t));
  half_t* f2wh = (half_t*)carve((size_t)4 * 128 * 512 * sizeof(half_t));
  float*  feat = (float*) carve((size_t)16 * 8192 * sizeof(float));

  // --- prep: fold conv/BN algebra, pad+transpose x, cast transformer weights to f16 ---
  k_prep_weff<<<(128 * 65 * 64) / 256, 256, 0, stream>>>(sw, bn1_g, bn2_g, tw1, tw2, tw3, tw4, Weff);
  k_prep_beta<<<1, 128, 0, stream>>>(sw, bn1_g, bn1_b, tb1, tb2, tb3, tb4, sb, bn2_g, bn2_b, beta);
  k_prep_x<<<(16 * 1072 * 64) / 256, 256, 0, stream>>>(x, xT);
  k_cast_f16<<<(4 * 128 * 128) / 256, 256, 0, stream>>>(wq, wqh, 4 * 128 * 128);
  k_cast_f16<<<(4 * 128 * 128) / 256, 256, 0, stream>>>(wk, wkh, 4 * 128 * 128);
  k_cast_f16<<<(4 * 128 * 128) / 256, 256, 0, stream>>>(wv, wvh, 4 * 128 * 128);
  k_cast_f16<<<(4 * 128 * 128) / 256, 256, 0, stream>>>(wo, woh, 4 * 128 * 128);
  k_cast_f16<<<(4 * 512 * 128) / 256, 256, 0, stream>>>(f1w, f1wh, 4 * 512 * 128);
  k_cast_f16<<<(4 * 128 * 512) / 256, 256, 0, stream>>>(f2w, f2wh, 4 * 128 * 512);

  // --- fused temporal+spatial conv + BN + ELU (WMMA), then pooling into token streams ---
  k_conv_wmma<<<1008, 256, 0, stream>>>(Weff, xT, beta, h);
  k_pool<<<(16 * 128 * 64) / 256, 256, 0, stream>>>(h, X);

  // --- 4 encoder layers on both streams stacked (M = 2*16*64 = 2048) ---
  for (int l = 0; l < 4; l++) {
    k_layernorm<<<256, 256, 0, stream>>>(X, ln1_g + l * 128, ln1_b + l * 128, H);
    k_gemm_wmma<<<128, 256, 0, stream>>>(H, wqh + l * 16384, bq + l * 128, 2048, 128, 128, Qf, nullptr, nullptr, 0);
    k_gemm_wmma<<<128, 256, 0, stream>>>(H, wkh + l * 16384, bk + l * 128, 2048, 128, 128, Kf, nullptr, nullptr, 0);
    k_gemm_wmma<<<128, 256, 0, stream>>>(H, wvh + l * 16384, bv + l * 128, 2048, 128, 128, Vf, nullptr, nullptr, 0);
    k_attention<<<256, 64, 0, stream>>>(Qf, Kf, Vf, Oh);
    k_gemm_wmma<<<128, 256, 0, stream>>>(Oh, woh + l * 16384, bo + l * 128, 2048, 128, 128, X, nullptr, X, 1);
    k_layernorm<<<256, 256, 0, stream>>>(X, ln2_g + l * 128, ln2_b + l * 128, H);
    k_gemm_wmma<<<512, 256, 0, stream>>>(H, f1wh + l * 65536, f1b + l * 512, 2048, 512, 128, nullptr, F1h, nullptr, 3);
    k_gemm_wmma<<<128, 256, 0, stream>>>(F1h, f2wh + l * 65536, f2b + l * 128, 2048, 128, 512, X, nullptr, X, 1);
  }

  // --- merge conv + BN3 + ELU, classifier ---
  k_final_feat<<<(16 * 64 * 128) / 256, 256, 0, stream>>>(X, ew, eb, bn3_g, bn3_b, feat);
  k_classifier<<<64, 32, 0, stream>>>(feat, cw, cb, (float*)d_out);
}
